// encoderHead_15152644621123
// MI455X (gfx1250) — compile-verified, matrix operand order checked by reading, not run on hardware
//
#include <hip/hip_runtime.h>
#include <hip/hip_bf16.h>

// Problem constants (from the reference)
#define B_  8
#define S_  2048
#define E_  1024
#define DK_ 1024
#define BS_ (B_ * S_)   // 16384 token rows

typedef __attribute__((ext_vector_type(16))) __bf16 v16bf;
typedef __attribute__((ext_vector_type(8)))  float  v8f;

// ---------------------------------------------------------------- helpers ---

// f32 -> bf16 round-to-nearest-even
__device__ __forceinline__ unsigned short f2bf(float f) {
  unsigned int u = __float_as_uint(f);
  u += 0x7FFFu + ((u >> 16) & 1u);
  return (unsigned short)(u >> 16);
}

__device__ __forceinline__ v8f wmma_bf16(v16bf a, v16bf b, v8f c) {
  // 8 args: (neg_a, A, neg_b, B, c_mod, C, reuse_a, reuse_b)
  return __builtin_amdgcn_wmma_f32_16x16x32_bf16(
      false, a, false, b, (short)0, c, false, false);
}

// Load a 16x32 A-operand (lane = M row) or 32x16 B-operand (lane = N col)
// fragment from a row-major bf16 source: element (r, k) at src[r*stride + k].
// ISA 16-bit layout: lanes 0-15 carry K = {0..7, 16..23}, lanes 16-31 add 8.
// Consecutive fragment elements pair into aligned dword loads.
__device__ __forceinline__ v16bf load_frag_bf16(const unsigned short* src,
                                                int stride, int lane) {
  union { v16bf v; unsigned int u[8]; } f;
  const unsigned short* p = src + (lane & 15) * stride + ((lane >> 4) << 3);
#pragma unroll
  for (int j = 0; j < 8; ++j) {
    const int k = ((j & 3) << 1) + ((j >> 2) << 4);   // 0,2,4,6,16,18,20,22
    f.u[j] = *(const unsigned int*)(p + k);
  }
  return f.v;
}

// Same fragment shape, but converting f32 source (LDS probabilities) -> bf16.
__device__ __forceinline__ v16bf load_frag_f32(const float* src,
                                               int stride, int lane) {
  union { v16bf v; unsigned short s[16]; } f;
  const float* p = src + (lane & 15) * stride + ((lane >> 4) << 3);
#pragma unroll
  for (int i = 0; i < 16; ++i) {
    const int k = (i & 7) + ((i >> 3) << 4);
    f.s[i] = f2bf(p[k]);
  }
  return f.v;
}

// ---------------------------------------------------------- mask / lengths ---

__global__ void zerolen_kernel(int* lengths) {
  if (threadIdx.x < B_) lengths[threadIdx.x] = 0;
}

// One wave per token row: valid iff any of its E elements is nonzero.
__global__ __launch_bounds__(256)
void rowmask_kernel(const float* __restrict__ x,
                    unsigned char* __restrict__ rowvalid,
                    int* __restrict__ lengths) {
  const int w = threadIdx.x >> 5, lane = threadIdx.x & 31;
  const int row = blockIdx.x * 8 + w;
  const float* p = x + (size_t)row * E_;
  int any = 0;
  for (int i = lane; i < E_; i += 32) any |= (p[i] != 0.0f);
  const unsigned long long bal = __ballot(any);
  if (lane == 0) {
    const int v = (bal != 0ull) ? 1 : 0;
    rowvalid[row] = (unsigned char)v;
    if (v) atomicAdd(lengths + (row / S_), 1);
  }
}

// ------------------------------------------------------------- QKV GEMM ----
// C[m,n] = sum_e X[m,e] * W[n,e]   (torch Linear: x @ W^T)
// 128x128 block tile, K-chunks of 32, double-buffered bf16 LDS staging,
// 8 waves in a 2(M) x 4(N) grid, each wave owns 64x32 = 4x2 WMMA tiles.
// z selects Wq/Wk/Wv; V is stored transposed (stride swap, branch-free).

// Stage one 128x32 f32 tile -> bf16 LDS buffer (row stride E_ in source).
__device__ __forceinline__ void stage_tile(const float* __restrict__ src,
                                           unsigned short* __restrict__ dst,
                                           int tid) {
  const int r0 = tid >> 3;            // 0..31
  const int c4 = (tid & 7) << 2;      // 0..28
#pragma unroll
  for (int pass = 0; pass < 4; ++pass) {
    const int row = pass * 32 + r0;
    const float4 v = *(const float4*)(src + (size_t)row * E_ + c4);
    dst[row * 32 + c4 + 0] = f2bf(v.x);
    dst[row * 32 + c4 + 1] = f2bf(v.y);
    dst[row * 32 + c4 + 2] = f2bf(v.z);
    dst[row * 32 + c4 + 3] = f2bf(v.w);
  }
}

__global__ __launch_bounds__(256)
void qkv_kernel(const float* __restrict__ x,
                const float* __restrict__ Wq,
                const float* __restrict__ Wk,
                const float* __restrict__ Wv,
                unsigned short* __restrict__ Qb,
                unsigned short* __restrict__ Kb,
                unsigned short* __restrict__ Vt) {
  __shared__ unsigned short As[2][128 * 32];   // x tile, bf16 (double buffer)
  __shared__ unsigned short Bs[2][128 * 32];   // W tile, bf16 (double buffer)
  const int m0 = blockIdx.x * 128;
  const int n0 = blockIdx.y * 128;
  const int z  = blockIdx.z;
  const float* Wp = (z == 0) ? Wq : (z == 1) ? Wk : Wv;

  // Branch-free output addressing: Q/K row-major, V transposed = stride swap.
  unsigned short* outp = (z == 0) ? Qb : (z == 1) ? Kb : Vt;
  const size_t mstride = (z == 2) ? (size_t)1   : (size_t)DK_;
  const size_t nstride = (z == 2) ? (size_t)BS_ : (size_t)1;

  const int tid  = threadIdx.x;
  const int w    = tid >> 5, lane = tid & 31;
  const int wm   = w & 1, wn = w >> 1;        // 2 (M) x 4 (N) wave grid

  const v8f vzero = {0.f, 0.f, 0.f, 0.f, 0.f, 0.f, 0.f, 0.f};
  v8f acc[4][2];
#pragma unroll
  for (int i = 0; i < 4; ++i)
#pragma unroll
    for (int j = 0; j < 2; ++j) acc[i][j] = vzero;

  const float* xb = x  + (size_t)m0 * E_;
  const float* wb = Wp + (size_t)n0 * E_;

  stage_tile(xb, As[0], tid);
  stage_tile(wb, Bs[0], tid);
  __syncthreads();

  const int NSTEP = E_ / 32;
  for (int step = 0; step < NSTEP; ++step) {
    const int cur = step & 1;
    if (step + 1 < NSTEP) {               // prefetch next chunk into other buf
      stage_tile(xb + (step + 1) * 32, As[cur ^ 1], tid);
      stage_tile(wb + (step + 1) * 32, Bs[cur ^ 1], tid);
    }
    v16bf bfr[2];
#pragma unroll
    for (int nt = 0; nt < 2; ++nt)
      bfr[nt] = load_frag_bf16(Bs[cur] + (wn * 32 + nt * 16) * 32, 32, lane);
#pragma unroll
    for (int mt = 0; mt < 4; ++mt) {
      const v16bf afr = load_frag_bf16(As[cur] + (wm * 64 + mt * 16) * 32, 32, lane);
#pragma unroll
      for (int nt = 0; nt < 2; ++nt)
        acc[mt][nt] = wmma_bf16(afr, bfr[nt], acc[mt][nt]);
    }
    __syncthreads();
  }

  // Store bf16 results. C/D layout: lane = N (0-15), VGPR j = M (lanes>=16: M+8)
  const int laneoff = (lane >> 4) << 3;
  const int nl = lane & 15;
#pragma unroll
  for (int mt = 0; mt < 4; ++mt)
#pragma unroll
    for (int nt = 0; nt < 2; ++nt)
#pragma unroll
      for (int j = 0; j < 8; ++j) {
        const size_t m = (size_t)(m0 + wm * 64 + mt * 16 + laneoff + j);
        const size_t n = (size_t)(n0 + wn * 32 + nt * 16 + nl);
        outp[m * mstride + n * nstride] = f2bf(acc[mt][nt][j]);
      }
}

// ------------------------------------------------------------- attention ---
// One block per (batch, 16-query tile). Full 16 x 2048 energy block held in
// LDS (f32), so softmax is exact two-pass with no HBM spill (320KB WGP LDS).
#define ATTN_SMEM (16 * S_ * 4 + 16 * DK_ * 2)   // 128KB energy + 32KB Q tile

__global__ __launch_bounds__(256)
void attn_kernel(const unsigned short* __restrict__ Qb,
                 const unsigned short* __restrict__ Kb,
                 const unsigned short* __restrict__ Vt,
                 const unsigned char* __restrict__ rowvalid,
                 const int* __restrict__ lengths,
                 float* __restrict__ out) {
  extern __shared__ char smem[];
  float*          eng = (float*)smem;                          // [16][S_]
  unsigned short* qs  = (unsigned short*)(smem + 16 * S_ * 4); // [16][DK_]

  const int q0  = blockIdx.x * 16;
  const int b   = blockIdx.y;
  const int bS  = b * S_;
  const int tid = threadIdx.x;
  const int w = tid >> 5, lane = tid & 31;
  const int len = lengths[b];
  const int laneoff = (lane >> 4) << 3;
  const int nl = lane & 15;
  const v8f vzero = {0.f, 0.f, 0.f, 0.f, 0.f, 0.f, 0.f, 0.f};

  // Stage Q tile: 16 contiguous rows -> flat 32KB LDS copy (dword loads).
  {
    const unsigned int* src = (const unsigned int*)(Qb + (size_t)(bS + q0) * DK_);
    unsigned int* dst = (unsigned int*)qs;
    for (int i = tid; i < 16 * DK_ / 2; i += 256) dst[i] = src[i];
  }
  __syncthreads();

  // energy = (Q K^T) / sqrt(DK), with -inf column mask for k >= len.
  // Each wave owns a PAIR of 16-key tiles per pass: two independent WMMA
  // accumulation chains (hides XDL RAW latency), one shared A-fragment.
  for (int tp = w; tp < S_ / 32; tp += 8) {            // wave-uniform loop
    v8f acc0 = vzero, acc1 = vzero;
    const unsigned short* kbase0 = Kb + (size_t)(bS + tp * 32) * DK_;
    const unsigned short* kbase1 = kbase0 + (size_t)16 * DK_;
    for (int e0 = 0; e0 < DK_; e0 += 32) {
      __builtin_prefetch((const void*)(kbase0 + e0 + 128), 0, 1); // stream K
      const v16bf a   = load_frag_bf16(qs + e0, DK_, lane);       // LDS
      const v16bf bf0 = load_frag_bf16(kbase0 + e0, DK_, lane);   // global
      const v16bf bf1 = load_frag_bf16(kbase1 + e0, DK_, lane);
      acc0 = wmma_bf16(a, bf0, acc0);
      acc1 = wmma_bf16(a, bf1, acc1);
    }
    const int kcol0 = tp * 32 + nl;
    const int kcol1 = kcol0 + 16;
    const bool dead0 = (kcol0 >= len);
    const bool dead1 = (kcol1 >= len);
#pragma unroll
    for (int j = 0; j < 8; ++j) {
      const float v0 = acc0[j] * 0.03125f;   // 1/sqrt(1024)
      const float v1 = acc1[j] * 0.03125f;
      eng[(laneoff + j) * S_ + kcol0] = dead0 ? -__builtin_inff() : v0;
      eng[(laneoff + j) * S_ + kcol1] = dead1 ? -__builtin_inff() : v1;
    }
  }
  __syncthreads();

  // Exact softmax over 2048 columns; each wave owns two query rows.
#pragma unroll
  for (int rr = 0; rr < 2; ++rr) {
    const int row = w * 2 + rr;
    float* er = eng + row * S_;
    float m = -__builtin_inff();
    for (int i = lane; i < S_; i += 32) m = fmaxf(m, er[i]);
#pragma unroll
    for (int o = 16; o > 0; o >>= 1) m = fmaxf(m, __shfl_xor(m, o));
    float sum = 0.f;
    for (int i = lane; i < S_; i += 32) {
      const float p = __expf(er[i] - m);   // exp(-inf - m) == 0 for masked
      er[i] = p;
      sum += p;
    }
#pragma unroll
    for (int o = 16; o > 0; o >>= 1) sum += __shfl_xor(sum, o);
    const float inv = 1.0f / sum;
    for (int i = lane; i < S_; i += 32) er[i] *= inv;
  }
  __syncthreads();

  // O = P · V. Wave w owns d-columns [w*128, w*128+128): 8 independent WMMA
  // accumulators sharing each P fragment.
  const int dn0 = w * 128;
  v8f oacc[8];
#pragma unroll
  for (int nt = 0; nt < 8; ++nt) oacc[nt] = vzero;

  for (int kt = 0; kt < S_; kt += 32) {
    const v16bf a = load_frag_f32(eng + kt, S_, lane);   // P from LDS -> bf16
#pragma unroll
    for (int nt = 0; nt < 8; ++nt) {
      // V is stored transposed: key index contiguous per d -> dword loads.
      const unsigned short* vb =
          Vt + (size_t)(dn0 + nt * 16) * BS_ + bS + kt;
      __builtin_prefetch((const void*)(vb + 32), 0, 1);  // stream V^T
      const v16bf bf = load_frag_bf16(vb, BS_, lane);
      oacc[nt] = wmma_bf16(a, bf, oacc[nt]);
    }
  }

  // Store f32 output, scaled by the query-row validity mask.
#pragma unroll
  for (int nt = 0; nt < 8; ++nt)
#pragma unroll
    for (int j = 0; j < 8; ++j) {
      const int qrow = laneoff + j;
      const float mk = rowvalid[bS + q0 + qrow] ? 1.0f : 0.0f;
      out[(size_t)(bS + q0 + qrow) * DK_ + dn0 + nt * 16 + nl] = oacc[nt][j] * mk;
    }
}

// ---------------------------------------------------------------- launch ---

extern "C" void kernel_launch(void* const* d_in, const int* in_sizes, int n_in,
                              void* d_out, int out_size, void* d_ws, size_t ws_size,
                              hipStream_t stream) {
  const float* x  = (const float*)d_in[0];
  const float* Wq = (const float*)d_in[1];
  const float* Wk = (const float*)d_in[2];
  const float* Wv = (const float*)d_in[3];
  float* out = (float*)d_out;

  // Workspace layout: Q, K (row-major) and V (transposed) in bf16, then
  // per-row validity bytes and per-batch lengths. Total ~101 MB.
  unsigned short* Qb = (unsigned short*)d_ws;
  unsigned short* Kb = Qb + (size_t)BS_ * DK_;
  unsigned short* Vt = Kb + (size_t)BS_ * DK_;
  unsigned char* rowvalid = (unsigned char*)(Vt + (size_t)BS_ * DK_);
  int* lengths = (int*)(rowvalid + BS_);

  (void)in_sizes; (void)n_in; (void)out_size; (void)ws_size;

  hipFuncSetAttribute(reinterpret_cast<const void*>(attn_kernel),
                      hipFuncAttributeMaxDynamicSharedMemorySize, ATTN_SMEM);

  zerolen_kernel<<<1, 32, 0, stream>>>(lengths);
  rowmask_kernel<<<BS_ / 8, 256, 0, stream>>>(x, rowvalid, lengths);
  qkv_kernel<<<dim3(BS_ / 128, DK_ / 128, 3), 256, 0, stream>>>(
      x, Wq, Wk, Wv, Qb, Kb, Vt);
  attn_kernel<<<dim3(S_ / 16, B_), 256, ATTN_SMEM, stream>>>(
      Qb, Kb, Vt, rowvalid, lengths, out);
}
